// baseline_30820685316246
// MI455X (gfx1250) — compile-verified, matrix operand order checked by reading, not run on hardware
//
#include <hip/hip_runtime.h>
#include <math.h>

#define NN     10000
#define NG     32
#define DIMF   512
#define NHEADS 4
#define HD     128
#define NE     160000
#define NE2    (NE + NN)
#define EPSBN  1e-5f

typedef __attribute__((ext_vector_type(16))) __bf16 v16bf;
typedef __attribute__((ext_vector_type(8)))  __bf16 v8bf;
typedef __attribute__((ext_vector_type(8)))  float  v8f;

// ---------- helpers ----------
__device__ __forceinline__ unsigned f2ord(float f) {
  unsigned b = __float_as_uint(f);
  return (b & 0x80000000u) ? ~b : (b | 0x80000000u);
}
__device__ __forceinline__ float ord2f(unsigned u) {
  unsigned b = (u & 0x80000000u) ? (u ^ 0x80000000u) : ~u;
  return __uint_as_float(b);
}
__device__ __forceinline__ v16bf load_bfrag(const __bf16* __restrict__ Br) {
  v8bf b0 = *(const v8bf*)(Br);
  v8bf b1 = *(const v8bf*)(Br + 16);
  v16bf b;
#pragma unroll
  for (int e = 0; e < 8; ++e) { b[e] = b0[e]; b[8 + e] = b1[e]; }
  return b;
}

// ---------- elementwise ----------
__global__ void k_fill_f32(float* p, float v, long n) {
  long i = (long)blockIdx.x * blockDim.x + threadIdx.x;
  long st = (long)gridDim.x * blockDim.x;
  for (; i < n; i += st) p[i] = v;
}
__global__ void k_fill_u32(unsigned* p, unsigned v, long n) {
  long i = (long)blockIdx.x * blockDim.x + threadIdx.x;
  long st = (long)gridDim.x * blockDim.x;
  for (; i < n; i += st) p[i] = v;
}
__global__ void k_copy_f32(const float* __restrict__ a, float* __restrict__ b, long n) {
  long i = (long)blockIdx.x * blockDim.x + threadIdx.x;
  long st = (long)gridDim.x * blockDim.x;
  for (; i < n; i += st) b[i] = a[i];
}
// x[NN,20] -> xp[NN,32] zero padded
__global__ void k_pad20(const float* __restrict__ x, float* __restrict__ xp, long n) {
  long i = (long)blockIdx.x * blockDim.x + threadIdx.x;
  long st = (long)gridDim.x * blockDim.x;
  for (; i < n; i += st) {
    long node = i >> 5; int c = (int)(i & 31);
    xp[i] = (c < 20) ? x[node * 20 + c] : 0.0f;
  }
}
// out[dst[e]*strideOut + c] += x[src[e]*dimIn + c]
__global__ void k_scatter(const float* __restrict__ x, const int* __restrict__ src,
                          const int* __restrict__ dst, float* __restrict__ out,
                          long total, int dimIn, int strideOut) {
  long i = (long)blockIdx.x * blockDim.x + threadIdx.x;
  long st = (long)gridDim.x * blockDim.x;
  for (; i < total; i += st) {
    long e = i / dimIn; int c = (int)(i % dimIn);
    atomicAdd(&out[(long)dst[e] * strideOut + c], x[(long)src[e] * dimIn + c]);
  }
}
// W[K,N] f32 -> Wt[N,Kp] bf16 (transposed, K zero-padded to Kp)
__global__ void k_prep_w(const float* __restrict__ W, __bf16* __restrict__ Wt,
                         int K, int N, int Kp, long total) {
  long i = (long)blockIdx.x * blockDim.x + threadIdx.x;
  long st = (long)gridDim.x * blockDim.x;
  for (; i < total; i += st) {
    long n = i / Kp; int k = (int)(i % Kp);
    Wt[i] = (k < K) ? (__bf16)W[(long)k * N + n] : (__bf16)0.0f;
  }
}

// ---------- WMMA GEMM: C[M,N] = A[M,lda(f32)] * Wt[N,Kp(bf16)]^T + bias ----------
// One wave produces a 16x64 strip: one A fragment feeds 4 WMMAs per K-step.
// Requires N % 64 == 0 (all Ns here are 512 or 128).
#define WMMA_BF16(ACC, AF, BF) \
  (ACC) = __builtin_amdgcn_wmma_f32_16x16x32_bf16(false, (AF), false, (BF), (short)0, (ACC), false, false)

__global__ void k_gemm(const float* __restrict__ A, int lda,
                       const __bf16* __restrict__ Wt, const float* __restrict__ bias,
                       int hasBias, float* __restrict__ C,
                       int M, int N, int Kp, int Mtiles, int nTiles, int relu) {
  int wave = threadIdx.x >> 5;
  int lane = threadIdx.x & 31;
  int tile = blockIdx.x * 8 + wave;
  if (tile >= nTiles) return;                 // wave-uniform exit, EXEC stays all-ones
  int mt = tile % Mtiles, ng = tile / Mtiles; // ng = 64-wide N group
  int half = lane >> 4, l16 = lane & 15;
  int row = mt * 16 + l16; if (row >= M) row = M - 1;
  const float*  Abase = A  + (long)row * lda + half * 8;
  const __bf16* Bbase = Wt + (long)(ng * 64 + l16) * Kp + half * 8;
  const long bstep = 16L * Kp;                // next 16-column tile of B

  v8f acc0 = {}, acc1 = {}, acc2 = {}, acc3 = {};
  for (int k0 = 0; k0 < Kp; k0 += 32) {
    const float*  Ar = Abase + k0;
    const __bf16* Br = Bbase + k0;
    if (k0 + 32 < Kp) {                       // prefetch next K-slice (global_prefetch_b8)
      __builtin_prefetch((const void*)(Ar + 32), 0, 0);
      __builtin_prefetch((const void*)(Br + 32), 0, 0);
    }
    float4 a0 = *(const float4*)(Ar);
    float4 a1 = *(const float4*)(Ar + 4);
    float4 a2 = *(const float4*)(Ar + 16);
    float4 a3 = *(const float4*)(Ar + 20);

    v16bf bf0 = load_bfrag(Br);
    v16bf bf1 = load_bfrag(Br + bstep);
    v16bf bf2 = load_bfrag(Br + 2 * bstep);
    v16bf bf3 = load_bfrag(Br + 3 * bstep);

    v16bf a;
    a[0]=(__bf16)a0.x; a[1]=(__bf16)a0.y; a[2]=(__bf16)a0.z; a[3]=(__bf16)a0.w;
    a[4]=(__bf16)a1.x; a[5]=(__bf16)a1.y; a[6]=(__bf16)a1.z; a[7]=(__bf16)a1.w;
    a[8]=(__bf16)a2.x; a[9]=(__bf16)a2.y; a[10]=(__bf16)a2.z; a[11]=(__bf16)a2.w;
    a[12]=(__bf16)a3.x; a[13]=(__bf16)a3.y; a[14]=(__bf16)a3.z; a[15]=(__bf16)a3.w;

    WMMA_BF16(acc0, a, bf0);
    WMMA_BF16(acc1, a, bf1);
    WMMA_BF16(acc2, a, bf2);
    WMMA_BF16(acc3, a, bf3);
  }

#pragma unroll
  for (int j = 0; j < 4; ++j) {
    v8f acc = (j == 0) ? acc0 : (j == 1) ? acc1 : (j == 2) ? acc2 : acc3;
    int col = ng * 64 + j * 16 + l16;
    float bv = hasBias ? bias[col] : 0.0f;
#pragma unroll
    for (int r = 0; r < 8; ++r) {
      int m = mt * 16 + r + half * 8;
      if (m < M) {
        float v = acc[r] + bv;
        if (relu) v = v > 0.0f ? v : 0.0f;
        C[(long)m * N + col] = v;
      }
    }
  }
}

// ---------- batch-norm stats / apply ----------
__global__ void k_colstats(const float* __restrict__ X, float* __restrict__ sum,
                           float* __restrict__ sumsq, int M, int N) {
  __shared__ float s1[256], s2[256];
  int col = blockIdx.x;
  float a = 0.f, b = 0.f;
  for (int r = threadIdx.x; r < M; r += 256) {
    float v = X[(long)r * N + col]; a += v; b += v * v;
  }
  s1[threadIdx.x] = a; s2[threadIdx.x] = b; __syncthreads();
  for (int s = 128; s > 0; s >>= 1) {
    if ((int)threadIdx.x < s) { s1[threadIdx.x] += s1[threadIdx.x + s]; s2[threadIdx.x] += s2[threadIdx.x + s]; }
    __syncthreads();
  }
  if (threadIdx.x == 0) { sum[col] = s1[0]; sumsq[col] = s2[0]; }
}
__global__ void k_bn_relu(const float* __restrict__ X, float* __restrict__ Y,
                          const float* __restrict__ sum, const float* __restrict__ sumsq,
                          const float* __restrict__ g, const float* __restrict__ b,
                          int M, int N, int relu, long total) {
  long i = (long)blockIdx.x * blockDim.x + threadIdx.x;
  long st = (long)gridDim.x * blockDim.x;
  float invM = 1.0f / (float)M;
  for (; i < total; i += st) {
    int c = (int)(i % N);
    float mu = sum[c] * invM;
    float var = sumsq[c] * invM - mu * mu;
    float v = g[c] * (X[i] - mu) * rsqrtf(var + EPSBN) + b[c];
    if (relu) v = fmaxf(v, 0.0f);
    Y[i] = v;
  }
}

// ---------- GAT ----------
__global__ void k_gat_alpha(const float* __restrict__ h, const float* __restrict__ asrc,
                            const float* __restrict__ adst, float* __restrict__ als,
                            float* __restrict__ ald, long total) {
  long i = (long)blockIdx.x * blockDim.x + threadIdx.x;
  long st = (long)gridDim.x * blockDim.x;
  for (; i < total; i += st) {
    long n = i >> 2; int hh = (int)(i & 3);
    const float* hp = h + n * DIMF + hh * HD;
    const float* as = asrc + hh * HD;
    const float* ad = adst + hh * HD;
    float s = 0.f, d = 0.f;
    for (int c = 0; c < HD; ++c) { float v = hp[c]; s += v * as[c]; d += v * ad[c]; }
    als[i] = s; ald[i] = d;
  }
}
__global__ void k_gat_pass1(const float* __restrict__ als, const float* __restrict__ ald,
                            const int* __restrict__ src, const int* __restrict__ dst,
                            float* __restrict__ ebuf, unsigned* __restrict__ mord, long total) {
  long i = (long)blockIdx.x * blockDim.x + threadIdx.x;
  long st = (long)gridDim.x * blockDim.x;
  for (; i < total; i += st) {
    long e = i >> 2; int hh = (int)(i & 3);
    int s, d;
    if (e < NE) { s = src[e]; d = dst[e]; } else { s = d = (int)(e - NE); }
    float v = als[s * 4 + hh] + ald[d * 4 + hh];
    v = v >= 0.f ? v : 0.2f * v;              // leaky_relu(0.2)
    ebuf[i] = v;
    atomicMax(&mord[d * 4 + hh], f2ord(v));   // float segment_max via ordered-uint
  }
}
__global__ void k_gat_pass2(float* __restrict__ ebuf, const unsigned* __restrict__ mord,
                            float* __restrict__ z, const int* __restrict__ dst, long total) {
  long i = (long)blockIdx.x * blockDim.x + threadIdx.x;
  long st = (long)gridDim.x * blockDim.x;
  for (; i < total; i += st) {
    long e = i >> 2; int hh = (int)(i & 3);
    int d = (e < NE) ? dst[e] : (int)(e - NE);
    float m = ord2f(mord[d * 4 + hh]);
    float ex = __expf(ebuf[i] - m);
    ebuf[i] = ex;
    atomicAdd(&z[d * 4 + hh], ex);
  }
}
__global__ void k_gat_pass3(const float* __restrict__ hmsg, const float* __restrict__ ebuf,
                            const float* __restrict__ z, const int* __restrict__ src,
                            const int* __restrict__ dst, float* __restrict__ out, long total) {
  long i = (long)blockIdx.x * blockDim.x + threadIdx.x;
  long st = (long)gridDim.x * blockDim.x;
  for (; i < total; i += st) {
    long e = i / DIMF; int c = (int)(i % DIMF); int hh = c >> 7;
    int s, d;
    if (e < NE) { s = src[e]; d = dst[e]; } else { s = d = (int)(e - NE); }
    float alpha = ebuf[e * 4 + hh] / (z[d * 4 + hh] + 1e-16f);
    atomicAdd(&out[(long)d * DIMF + c], hmsg[(long)s * DIMF + c] * alpha);
  }
}
__global__ void k_pool(const float* __restrict__ x, const float* __restrict__ bias,
                       const int* __restrict__ batch, float* __restrict__ g, long total) {
  long i = (long)blockIdx.x * blockDim.x + threadIdx.x;
  long st = (long)gridDim.x * blockDim.x;
  for (; i < total; i += st) {
    long n = i / DIMF; int c = (int)(i % DIMF);
    atomicAdd(&g[(long)batch[n] * DIMF + c], x[i] + bias[c]);
  }
}

// ---------- head ----------
__global__ void k_concat(const float* __restrict__ a, const float* __restrict__ b,
                         float* __restrict__ out, long total) {
  long i = (long)blockIdx.x * blockDim.x + threadIdx.x;
  long st = (long)gridDim.x * blockDim.x;
  for (; i < total; i += st) {
    long m = i / 256; int j = (int)(i % 256);
    out[i] = (j < 128) ? a[m * 128 + j] : b[m * 128 + (j - 128)];
  }
}
__global__ void k_reg2(const float* __restrict__ h, const float* __restrict__ W,
                       const float* __restrict__ b, float* __restrict__ out) {
  int t = threadIdx.x;
  if (t < NG) {
    float s = 0.f;
    for (int k = 0; k < 128; ++k) s += h[t * 128 + k] * W[k];
    out[t] = s + b[0];
  }
}

// ---------- host orchestration ----------
static inline int nblk(long n) { long b = (n + 255) / 256; return (int)(b > 49152 ? 49152 : b); }

static void gemm(hipStream_t st, const float* A, int lda, const __bf16* Wt,
                 const float* bias, int hasBias, float* C, int M, int N, int Kp, int relu) {
  int Mt = (M + 15) / 16, Ngr = N / 64;       // N % 64 == 0 for all layers here
  int tiles = Mt * Ngr;
  k_gemm<<<(tiles + 7) / 8, 256, 0, st>>>(A, lda, Wt, bias, hasBias, C, M, N, Kp, Mt, tiles, relu);
}

struct WS {
  float *xpad, *bufA, *bufB, *bufC;
  __bf16 *w1t, *w2t, *w3t, *w4t, *w5t, *w6t, *w7t, *w8t;
  float *csum, *csq;
  float *als, *ald, *ebuf, *zb; unsigned* mord;
  float *gpool, *fc1, *fc1a, *bout0, *bout1, *gcat, *regh;
};

// param indices within a branch block (dict order of branch())
// 0 gin1_W1, 1 gin1_b1, 2 gin1_W2, 3 gin1_b2, 4 bn1_g, 5 bn1_b,
// 6 gin2_W1, 7 gin2_b1, 8 gin2_W2, 9 gin2_b2, 10 bn2_g, 11 bn2_b,
// 12 gat_W, 13 gat_asrc, 14 gat_adst, 15 gat_bias,
// 16 fc_W1, 17 fc_b1, 18 fc_bn_g, 19 fc_bn_b, 20 fc_W2, 21 fc_b2
static void run_branch(hipStream_t st, const WS& w, const float* x, const int* src,
                       const int* dst, const int* batch, void* const* P, float* bout) {
#define PF(i) ((const float*)P[(i)])
  // weight prep (bf16, transposed, K padded)
  k_prep_w<<<nblk(512L * 32), 256, 0, st>>>(PF(0),  w.w1t, 20,   DIMF, 32,   512L * 32);
  k_prep_w<<<nblk(512L * 512),256, 0, st>>>(PF(2),  w.w2t, DIMF, DIMF, DIMF, 512L * 512);
  k_prep_w<<<nblk(512L * 512),256, 0, st>>>(PF(6),  w.w3t, DIMF, DIMF, DIMF, 512L * 512);
  k_prep_w<<<nblk(512L * 512),256, 0, st>>>(PF(8),  w.w4t, DIMF, DIMF, DIMF, 512L * 512);
  k_prep_w<<<nblk(512L * 512),256, 0, st>>>(PF(12), w.w5t, DIMF, DIMF, DIMF, 512L * 512);
  k_prep_w<<<nblk(512L * 512),256, 0, st>>>(PF(16), w.w6t, DIMF, DIMF, DIMF, 512L * 512);
  k_prep_w<<<nblk(128L * 512),256, 0, st>>>(PF(20), w.w7t, DIMF, 128,  DIMF, 128L * 512);

  // GIN1: A = x + segment_sum(x[src], dst)  (padded 20 -> 32)
  k_pad20<<<nblk((long)NN * 32), 256, 0, st>>>(x, w.xpad, (long)NN * 32);
  k_scatter<<<nblk((long)NE * 20), 256, 0, st>>>(x, src, dst, w.xpad, (long)NE * 20, 20, 32);
  gemm(st, w.xpad, 32, w.w1t, PF(1), 1, w.bufB, NN, DIMF, 32, 1);
  gemm(st, w.bufB, DIMF, w.w2t, PF(3), 1, w.bufC, NN, DIMF, DIMF, 0);
  k_colstats<<<DIMF, 256, 0, st>>>(w.bufC, w.csum, w.csq, NN, DIMF);
  k_bn_relu<<<nblk((long)NN * DIMF), 256, 0, st>>>(w.bufC, w.bufA, w.csum, w.csq,
                                                   PF(4), PF(5), NN, DIMF, 1, (long)NN * DIMF);
  // GIN2
  k_copy_f32<<<nblk((long)NN * DIMF), 256, 0, st>>>(w.bufA, w.bufC, (long)NN * DIMF);
  k_scatter<<<nblk((long)NE * DIMF), 256, 0, st>>>(w.bufA, src, dst, w.bufC, (long)NE * DIMF, DIMF, DIMF);
  gemm(st, w.bufC, DIMF, w.w3t, PF(7), 1, w.bufB, NN, DIMF, DIMF, 1);
  gemm(st, w.bufB, DIMF, w.w4t, PF(9), 1, w.bufC, NN, DIMF, DIMF, 0);
  k_colstats<<<DIMF, 256, 0, st>>>(w.bufC, w.csum, w.csq, NN, DIMF);
  k_bn_relu<<<nblk((long)NN * DIMF), 256, 0, st>>>(w.bufC, w.bufA, w.csum, w.csq,
                                                   PF(10), PF(11), NN, DIMF, 1, (long)NN * DIMF);
  // GAT
  gemm(st, w.bufA, DIMF, w.w5t, PF(15), 0, w.bufB, NN, DIMF, DIMF, 0); // h @ W
  k_gat_alpha<<<nblk((long)NN * 4), 256, 0, st>>>(w.bufB, PF(13), PF(14), w.als, w.ald, (long)NN * 4);
  k_fill_u32<<<nblk((long)NN * 4), 256, 0, st>>>(w.mord, 0u, (long)NN * 4);
  k_fill_f32<<<nblk((long)NN * 4), 256, 0, st>>>(w.zb, 0.f, (long)NN * 4);
  k_gat_pass1<<<nblk((long)NE2 * 4), 256, 0, st>>>(w.als, w.ald, src, dst, w.ebuf, w.mord, (long)NE2 * 4);
  k_gat_pass2<<<nblk((long)NE2 * 4), 256, 0, st>>>(w.ebuf, w.mord, w.zb, dst, (long)NE2 * 4);
  k_fill_f32<<<nblk((long)NN * DIMF), 256, 0, st>>>(w.bufC, 0.f, (long)NN * DIMF);
  k_gat_pass3<<<nblk((long)NE2 * DIMF), 256, 0, st>>>(w.bufB, w.ebuf, w.zb, src, dst, w.bufC, (long)NE2 * DIMF);
  // pool (+ gat bias folded per row)
  k_fill_f32<<<nblk((long)NG * DIMF), 256, 0, st>>>(w.gpool, 0.f, (long)NG * DIMF);
  k_pool<<<nblk((long)NN * DIMF), 256, 0, st>>>(w.bufC, PF(15), batch, w.gpool, (long)NN * DIMF);
  // fc head
  gemm(st, w.gpool, DIMF, w.w6t, PF(17), 1, w.fc1, NG, DIMF, DIMF, 0);
  k_colstats<<<DIMF, 256, 0, st>>>(w.fc1, w.csum, w.csq, NG, DIMF);
  k_bn_relu<<<nblk((long)NG * DIMF), 256, 0, st>>>(w.fc1, w.fc1a, w.csum, w.csq,
                                                   PF(18), PF(19), NG, DIMF, 1, (long)NG * DIMF);
  gemm(st, w.fc1a, DIMF, w.w7t, PF(21), 1, bout, NG, 128, DIMF, 0);
#undef PF
}

extern "C" void kernel_launch(void* const* d_in, const int* in_sizes, int n_in,
                              void* d_out, int out_size, void* d_ws, size_t ws_size,
                              hipStream_t stream) {
  (void)in_sizes; (void)n_in; (void)out_size; (void)ws_size;
  char* base = (char*)d_ws;
  size_t off = 0;
  auto alloc = [&](size_t bytes) -> void* {
    void* p = base + off;
    off += (bytes + 255) & ~(size_t)255;
    return p;
  };
  WS w;
  w.xpad  = (float*)alloc((size_t)NN * 32 * 4);
  w.bufA  = (float*)alloc((size_t)NN * DIMF * 4);
  w.bufB  = (float*)alloc((size_t)NN * DIMF * 4);
  w.bufC  = (float*)alloc((size_t)NN * DIMF * 4);
  w.w1t   = (__bf16*)alloc((size_t)512 * 32 * 2);
  w.w2t   = (__bf16*)alloc((size_t)512 * 512 * 2);
  w.w3t   = (__bf16*)alloc((size_t)512 * 512 * 2);
  w.w4t   = (__bf16*)alloc((size_t)512 * 512 * 2);
  w.w5t   = (__bf16*)alloc((size_t)512 * 512 * 2);
  w.w6t   = (__bf16*)alloc((size_t)512 * 512 * 2);
  w.w7t   = (__bf16*)alloc((size_t)128 * 512 * 2);
  w.w8t   = (__bf16*)alloc((size_t)128 * 256 * 2);
  w.csum  = (float*)alloc((size_t)DIMF * 4);
  w.csq   = (float*)alloc((size_t)DIMF * 4);
  w.als   = (float*)alloc((size_t)NN * 4 * 4);
  w.ald   = (float*)alloc((size_t)NN * 4 * 4);
  w.ebuf  = (float*)alloc((size_t)NE2 * 4 * 4);
  w.zb    = (float*)alloc((size_t)NN * 4 * 4);
  w.mord  = (unsigned*)alloc((size_t)NN * 4 * 4);
  w.gpool = (float*)alloc((size_t)NG * DIMF * 4);
  w.fc1   = (float*)alloc((size_t)NG * DIMF * 4);
  w.fc1a  = (float*)alloc((size_t)NG * DIMF * 4);
  w.bout0 = (float*)alloc((size_t)NG * 128 * 4);
  w.bout1 = (float*)alloc((size_t)NG * 128 * 4);
  w.gcat  = (float*)alloc((size_t)NG * 256 * 4);
  w.regh  = (float*)alloc((size_t)NG * 128 * 4);

  const float* x_wl = (const float*)d_in[0];
  const int*   e_wl = (const int*)d_in[1];   // [2, NE]
  const int*   b_wl = (const int*)d_in[2];
  const float* x_mt = (const float*)d_in[3];
  const int*   e_mt = (const int*)d_in[4];
  const int*   b_mt = (const int*)d_in[5];
  void* const* Pwl = d_in + 6;               // 22 branch params
  void* const* Pmt = d_in + 28;
  const float* reg_W1 = (const float*)d_in[50];
  const float* reg_b1 = (const float*)d_in[51];
  const float* reg_W2 = (const float*)d_in[52];
  const float* reg_b2 = (const float*)d_in[53];

  run_branch(stream, w, x_wl, e_wl, e_wl + NE, b_wl, Pwl, w.bout0);
  run_branch(stream, w, x_mt, e_mt, e_mt + NE, b_mt, Pmt, w.bout1);

  k_concat<<<nblk((long)NG * 256), 256, 0, stream>>>(w.bout0, w.bout1, w.gcat, (long)NG * 256);
  k_prep_w<<<nblk(128L * 256), 256, 0, stream>>>(reg_W1, w.w8t, 256, 128, 256, 128L * 256);
  gemm(stream, w.gcat, 256, w.w8t, reg_b1, 1, w.regh, NG, 128, 256, 1);
  k_reg2<<<1, 32, 0, stream>>>(w.regh, reg_W2, reg_b2, (float*)d_out);
}
